// TemplateStack_14542759264357
// MI455X (gfx1250) — matrix operand order repaired and checked by reading, compiled, not run on hardware
//
#include <hip/hip_runtime.h>
#include <hip/hip_bf16.h>
#include <math.h>

// ---------------------------------------------------------------------------
// AF2 Template embedder for gfx1250 (MI455X).
//  * All matmuls via v_wmma_f32_16x16x32_f16 (wave32).
//  * Global->LDS tile staging via CDNA5 async-LDS engine
//    (global_load_async_to_lds_b128/b32 + s_wait_asynccnt).
//  * f32 master data; f16 fragments built from LDS per the CDNA5 VGPR layout
//    tables (CDNA5 ISA §7.12.2).
// ---------------------------------------------------------------------------

typedef __attribute__((ext_vector_type(16))) _Float16 v16h;
typedef __attribute__((ext_vector_type(8)))  float    v8f;

static constexpr int LDIM = 128;   // L
static constexpr int NT   = 4;     // B*T templates
static constexpr int CT   = 64;    // c_t
static constexpr int NH   = 4;     // heads
static constexpr int CA   = 16;    // c_att

// ---- CDNA5 async global->LDS copies (ASYNCcnt) ----------------------------
// Generic LDS pointers carry the LDS byte offset in their low 32 bits
// (aperture layout, CDNA5 ISA), so truncation gives the
// ds-relative address the async ops want.
__device__ __forceinline__ void async_b128(void* lds, const float* g) {
    unsigned off = (unsigned)(uintptr_t)lds;
    asm volatile("global_load_async_to_lds_b128 %0, %1, off"
                 :: "v"(off), "v"(g) : "memory");
}
__device__ __forceinline__ void async_b32(void* lds, const float* g) {
    unsigned off = (unsigned)(uintptr_t)lds;
    asm volatile("global_load_async_to_lds_b32 %0, %1, off"
                 :: "v"(off), "v"(g) : "memory");
}
__device__ __forceinline__ void wait_async0() {
    asm volatile("s_wait_asynccnt 0" ::: "memory");
}

// ============================================================================
// Generic row-major GEMM:  Y[r,n] = epilogue( sum_k X[r,k] * W[k,n] )
// Block = 128 threads (4 waves): 16 rows x 64 cols; wave w owns cols w*16..+16.
// A tile 16x32 staged once per block per K-chunk; B panel 32x64.
// ============================================================================
__global__ void __launch_bounds__(128)
gemm_f32_wmma_kernel(const float* __restrict__ X, int ldx,
                     const float* __restrict__ W, int ldw,
                     const float* __restrict__ bias,
                     float* __restrict__ Y, int ldy,
                     int R, int K, int N,
                     int act,                               // 0 none, 1 relu, 2 sigmoid
                     const float* __restrict__ rowscale,    // per-row scalar (may be null)
                     const float* __restrict__ ewmul, int ldew, // elementwise mul (may be null)
                     int accumulate,                        // Y += result
                     int transpose_store, int Ld)           // remap row (n,i,j)->(n,j,i)
{
    __shared__ __align__(16) float sA[16][32];
    __shared__ __align__(16) float sB[32][64];

    const int colblocks = (N + 63) >> 6;
    const int rt  = blockIdx.x / colblocks;
    const int cb  = blockIdx.x % colblocks;
    const int r0  = rt << 4;
    const int nb0 = cb << 6;
    const int tid  = threadIdx.x;
    const int wave = tid >> 5;
    const int lane = tid & 31;
    const int n0   = nb0 + (wave << 4);

    const bool fullK = (K & 31) == 0;
    const bool asyncA = fullK && (r0 + 16 <= R) && ((ldx & 3) == 0);
    const bool asyncB = fullK && (nb0 + 64 <= N) && ((ldw & 3) == 0);

    v8f acc = {};
    for (int k0 = 0; k0 < K; k0 += 32) {
        // ---- stage A tile 16x32 ----
        if (asyncA) {
            int row = tid >> 3, col4 = (tid & 7) << 2;            // 128 thr * b128
            async_b128(&sA[row][col4], &X[(size_t)(r0 + row) * ldx + k0 + col4]);
            if (k0 + 32 < K)
                __builtin_prefetch(&X[(size_t)(r0 + row) * ldx + k0 + 32 + col4], 0, 3);
        } else {
            for (int idx = tid; idx < 512; idx += 128) {
                int r = idx >> 5, kk = idx & 31;
                int gr = r0 + r, gk = k0 + kk;
                sA[r][kk] = (gr < R && gk < K) ? X[(size_t)gr * ldx + gk] : 0.f;
            }
        }
        // ---- stage B panel 32x64 ----
        if (asyncB) {
            for (int q = tid; q < 512; q += 128) {                // 512 * b128
                int row = q >> 4, col4 = (q & 15) << 2;
                async_b128(&sB[row][col4], &W[(size_t)(k0 + row) * ldw + nb0 + col4]);
            }
        } else {
            for (int idx = tid; idx < 2048; idx += 128) {
                int kk = idx >> 6, c = idx & 63;
                int gk = k0 + kk, gc = nb0 + c;
                sB[kk][c] = (gk < K && gc < N) ? W[(size_t)gk * ldw + gc] : 0.f;
            }
        }
        wait_async0();
        __syncthreads();
        // A fragment (16x32 f16): lanes 0-15 hold K {0..7,16..23}, lanes 16-31 K {8..15,24..31}
        const int m    = lane & 15;
        const int ksel = (lane >> 4) << 3;   // 0 or 8
        const int kb   = (lane >> 4) << 4;   // 0 or 16
        v16h a, b;
#pragma unroll
        for (int e = 0; e < 8; ++e) {
            a[e]     = (_Float16)sA[m][ksel + e];
            a[8 + e] = (_Float16)sA[m][16 + ksel + e];
        }
#pragma unroll
        for (int e = 0; e < 16; ++e) b[e] = (_Float16)sB[((e >> 3) << 4) + ksel + (e & 7)][(wave << 4) + m];
        acc = __builtin_amdgcn_wmma_f32_16x16x32_f16(false, a, false, b,
                                                     (short)0, acc, false, false);
        __syncthreads();
    }
    // Epilogue.  D layout: VGPR r -> (M=r lanes 0-15, M=r+8 lanes 16-31), N = lane&15.
    const int nc   = n0 + (lane & 15);
    const int half = lane >> 4;
#pragma unroll
    for (int r = 0; r < 8; ++r) {
        int gr = r0 + r + half * 8;
        if (gr >= R || nc >= N) continue;
        float y = acc[r];
        if (bias) y += bias[nc];
        if (act == 1) y = y > 0.f ? y : 0.f;
        else if (act == 2) y = 1.f / (1.f + __expf(-y));
        if (rowscale) y *= rowscale[gr];
        if (ewmul)    y *= ewmul[(size_t)gr * ldew + nc];
        size_t orow = (size_t)gr;
        if (transpose_store) {
            int n_ = gr / (Ld * Ld);
            int rem = gr - n_ * Ld * Ld;
            int i_ = rem / Ld, j_ = rem - i_ * Ld;
            orow = (size_t)(n_ * Ld + j_) * Ld + i_;
        }
        if (accumulate) Y[orow * ldy + nc] += y;
        else            Y[orow * ldy + nc]  = y;
    }
}

// ============================================================================
// Row LayerNorm (C = blockDim.x; 64 or 128), optional transposed (n,j,i) read.
// ============================================================================
__global__ void layernorm_kernel(const float* __restrict__ X,
                                 const float* __restrict__ gamma,
                                 const float* __restrict__ beta,
                                 float* __restrict__ Y,
                                 int C, int transpose_read, int Ld)
{
    __shared__ float red[128];
    const int r = blockIdx.x, tid = threadIdx.x;
    size_t srow = r;
    if (transpose_read) {
        int n_ = r / (Ld * Ld);
        int rem = r - n_ * Ld * Ld;
        int i_ = rem / Ld, j_ = rem - i_ * Ld;
        srow = (size_t)(n_ * Ld + j_) * Ld + i_;
    }
    float x = X[srow * C + tid];
    red[tid] = x; __syncthreads();
    for (int s = C >> 1; s > 0; s >>= 1) { if (tid < s) red[tid] += red[tid + s]; __syncthreads(); }
    float mu = red[0] / C; __syncthreads();
    float d = x - mu;
    red[tid] = d * d; __syncthreads();
    for (int s = C >> 1; s > 0; s >>= 1) { if (tid < s) red[tid] += red[tid + s]; __syncthreads(); }
    float var = red[0] / C;
    Y[(size_t)r * C + tid] = d * rsqrtf(var + 1e-5f) * gamma[tid] + beta[tid];
}

// pm[n,i,k] = tm[n,i] * tm[n,k]
__global__ void pair_mask_kernel(const float* __restrict__ tm, float* __restrict__ pm, int total)
{
    int idx = blockIdx.x * blockDim.x + threadIdx.x;
    if (idx >= total) return;
    int n = idx / (LDIM * LDIM);
    int rem = idx - n * LDIM * LDIM;
    int i = rem / LDIM, k = rem - i * LDIM;
    pm[idx] = tm[n * LDIM + i] * tm[n * LDIM + k];
}

// out = a * b * rowscale[row]
__global__ void ewmul_kernel(const float* __restrict__ a, const float* __restrict__ b,
                             const float* __restrict__ rowscale,
                             float* __restrict__ out, int C, size_t total)
{
    size_t i = (size_t)blockIdx.x * blockDim.x + threadIdx.x;
    if (i >= total) return;
    float v = a[i] * b[i];
    if (rowscale) v *= rowscale[i / C];
    out[i] = v;
}

// ============================================================================
// Triangle attention core. One wave per (n, i, h, j-tile).
//   S = Q_i K_i^T / 4 + tb[n,j,k,h] + (pm[n,i,k]-1)*1e9 ; softmax_k ; O = A V_i
// ============================================================================
__global__ void __launch_bounds__(32)
tri_attn_kernel(const float* __restrict__ Q, const float* __restrict__ K,
                const float* __restrict__ V, const float* __restrict__ TB,
                const float* __restrict__ PM, float* __restrict__ O)
{
    constexpr int JT = LDIM / 16;
    int bid = blockIdx.x;
    const int jt = bid % JT;  bid /= JT;
    const int h  = bid % NH;  bid /= NH;
    const int i  = bid % LDIM; bid /= LDIM;
    const int n  = bid;

    __shared__ __align__(16) float sKf[LDIM][16];
    __shared__ __align__(16) float sVf[LDIM][16];
    __shared__ __align__(16) float sQf[16][16];
    __shared__ float    sS[16][LDIM];
    __shared__ _Float16 sA16[16][LDIM];

    const int lane = threadIdx.x;
    const size_t rowbase = ((size_t)(n * LDIM + i)) * LDIM;   // rows (n,i,*)

    // async-LDS staging: each head slice is 16 floats = 4 x b128, 64B aligned.
    for (int idx = lane; idx < LDIM * 4; idx += 32) {          // 512 b128 for K and V
        int k = idx >> 2, c4 = (idx & 3) << 2;
        size_t off = (rowbase + k) * (NH * CA) + h * CA + c4;
        async_b128(&sKf[k][c4], &K[off]);
        async_b128(&sVf[k][c4], &V[off]);
    }
    for (int idx = lane; idx < 64; idx += 32) {
        int jr = idx >> 2, c4 = (idx & 3) << 2;
        async_b128(&sQf[jr][c4],
                   &Q[(rowbase + jt * 16 + jr) * (NH * CA) + h * CA + c4]);
    }
    wait_async0();
    __syncthreads();

    const int m    = lane & 15;
    const int ksel = (lane >> 4) << 3;
    const int kb   = (lane >> 4) << 4;
    const int half = lane >> 4;

    // ---- S = Q K^T  (K-dim 16 zero-padded to 32) ----
    v16h afr = {};
#pragma unroll
    for (int e = 0; e < 8; ++e) afr[e] = (_Float16)sQf[m][ksel + e]; // K 16..31 stay 0
    for (int ct = 0; ct < JT; ++ct) {
        v16h bfr;
#pragma unroll
        for (int e = 0; e < 16; ++e)
            bfr[e] = (e < 8) ? (_Float16)sKf[ct * 16 + m][ksel + e] : (_Float16)0.f;
        v8f accs = {};
        accs = __builtin_amdgcn_wmma_f32_16x16x32_f16(false, afr, false, bfr,
                                                      (short)0, accs, false, false);
#pragma unroll
        for (int r = 0; r < 8; ++r) sS[r + half * 8][ct * 16 + m] = accs[r];
    }
    __syncthreads();

    // ---- scale + triangle bias + key mask ----
    for (int idx = lane; idx < 16 * LDIM; idx += 32) {
        int r = idx >> 7, k = idx & (LDIM - 1);
        float v = sS[r][k] * 0.25f;                      // 1/sqrt(16)
        v += TB[((size_t)(n * LDIM + (jt * 16 + r)) * LDIM + k) * NH + h];
        v += (PM[rowbase + k] - 1.f) * 1e9f;
        sS[r][k] = v;
    }
    __syncthreads();

    // ---- softmax over k, per row ----
    for (int r = 0; r < 16; ++r) {
        float x0 = sS[r][lane], x1 = sS[r][lane + 32], x2 = sS[r][lane + 64], x3 = sS[r][lane + 96];
        float mx = fmaxf(fmaxf(x0, x1), fmaxf(x2, x3));
        for (int off = 16; off > 0; off >>= 1) mx = fmaxf(mx, __shfl_xor(mx, off, 32));
        float e0 = __expf(x0 - mx), e1 = __expf(x1 - mx), e2 = __expf(x2 - mx), e3 = __expf(x3 - mx);
        float sum = e0 + e1 + e2 + e3;
        for (int off = 16; off > 0; off >>= 1) sum += __shfl_xor(sum, off, 32);
        float inv = 1.f / sum;
        sA16[r][lane]      = (_Float16)(e0 * inv);
        sA16[r][lane + 32] = (_Float16)(e1 * inv);
        sA16[r][lane + 64] = (_Float16)(e2 * inv);
        sA16[r][lane + 96] = (_Float16)(e3 * inv);
    }
    __syncthreads();

    // ---- O = A (16x128) @ V (128x16), 4 K-chunks of 32 ----
    v8f acco = {};
    for (int kc = 0; kc < 4; ++kc) {
        v16h af, bf;
#pragma unroll
        for (int e = 0; e < 8; ++e) {
            af[e]     = sA16[m][kc * 32 + ksel + e];
            af[8 + e] = sA16[m][kc * 32 + 16 + ksel + e];
        }
#pragma unroll
        for (int e = 0; e < 16; ++e) bf[e] = (_Float16)sVf[kc * 32 + ((e >> 3) << 4) + ksel + (e & 7)][m];
        acco = __builtin_amdgcn_wmma_f32_16x16x32_f16(false, af, false, bf,
                                                      (short)0, acco, false, false);
    }
#pragma unroll
    for (int r = 0; r < 8; ++r) {
        int j = jt * 16 + r + half * 8;
        O[(rowbase + j) * (NH * CA) + h * CA + m] = acco[r];
    }
}

// ============================================================================
// Triangle multiplication einsum. One wave per (n, channel c, i-tile, j-tile).
//   outgoing: x[i,j,c] = sum_k a[i,k,c] b[j,k,c]
//   incoming: x[i,j,c] = sum_k a[k,i,c] b[k,j,c]
// Channel gathers are stride-64 floats: use async b32 loads into LDS.
// ============================================================================
__global__ void __launch_bounds__(32)
tri_mul_einsum_kernel(const float* __restrict__ A, const float* __restrict__ Bm,
                      float* __restrict__ X, int incoming)
{
    constexpr int TL = LDIM / 16;
    int bid = blockIdx.x;
    const int jt = bid % TL; bid /= TL;
    const int it = bid % TL; bid /= TL;
    const int c  = bid % CT; bid /= CT;
    const int n  = bid;

    __shared__ float sA[16][32];
    __shared__ float sB[32][16];
    const int lane = threadIdx.x;
    v8f acc = {};
    for (int k0 = 0; k0 < LDIM; k0 += 32) {
        for (int idx = lane; idx < 512; idx += 32) {
            int r = idx >> 5, kk = idx & 31;
            int gi = it * 16 + r, gk = k0 + kk;
            size_t srow = incoming ? ((size_t)(n * LDIM + gk) * LDIM + gi)
                                   : ((size_t)(n * LDIM + gi) * LDIM + gk);
            async_b32(&sA[r][kk], &A[srow * CT + c]);
        }
        for (int idx = lane; idx < 512; idx += 32) {
            int kk = idx >> 4, cc = idx & 15;
            int gj = jt * 16 + cc, gk = k0 + kk;
            size_t srow = incoming ? ((size_t)(n * LDIM + gk) * LDIM + gj)
                                   : ((size_t)(n * LDIM + gj) * LDIM + gk);
            async_b32(&sB[kk][cc], &Bm[srow * CT + c]);
        }
        wait_async0();
        __syncthreads();
        const int m = lane & 15, ksel = (lane >> 4) << 3, kb = (lane >> 4) << 4;
        v16h a, b;
#pragma unroll
        for (int e = 0; e < 8; ++e) {
            a[e]     = (_Float16)sA[m][ksel + e];
            a[8 + e] = (_Float16)sA[m][16 + ksel + e];
        }
#pragma unroll
        for (int e = 0; e < 16; ++e) b[e] = (_Float16)sB[((e >> 3) << 4) + ksel + (e & 7)][m];
        acc = __builtin_amdgcn_wmma_f32_16x16x32_f16(false, a, false, b,
                                                     (short)0, acc, false, false);
        __syncthreads();
    }
    const int m = lane & 15, half = lane >> 4;
#pragma unroll
    for (int r = 0; r < 8; ++r) {
        int gi = it * 16 + r + half * 8, gj = jt * 16 + m;
        X[((size_t)(n * LDIM + gi) * LDIM + gj) * CT + c] = acc[r];
    }
}

// ============================================================================
// Pointwise attention over templates (AF2 Alg 17). One block (64 thr) per (i,j).
// ============================================================================
__global__ void pwa_attn_kernel(const float* __restrict__ QZ,   // [L*L,64]
                                const float* __restrict__ KT,   // [T,L*L,64]
                                const float* __restrict__ VT,   // [T,L*L,64]
                                const float* __restrict__ tmask,// [T*L]
                                float* __restrict__ OUT)        // [L*L,64]
{
    const int ij = blockIdx.x;
    const int i = ij / LDIM, j = ij - i * LDIM;
    const int tid = threadIdx.x;          // h*16 + c
    const int h = tid >> 4;

    float logits[NT], pmv[NT];
    for (int t = 0; t < NT; ++t) {
        pmv[t] = tmask[t * LDIM + i] * tmask[t * LDIM + j];
        float s = 0.f;
        for (int c = 0; c < CA; ++c)
            s += QZ[(size_t)ij * 64 + h * CA + c] *
                 KT[((size_t)t * (LDIM * LDIM) + ij) * 64 + h * CA + c];
        logits[t] = (pmv[t] == 0.f) ? -1e9f : s * 0.25f;
    }
    float mx = logits[0];
    for (int t = 1; t < NT; ++t) mx = fmaxf(mx, logits[t]);
    float e[NT], sum = 0.f;
    for (int t = 0; t < NT; ++t) { e[t] = __expf(logits[t] - mx); sum += e[t]; }
    float s2 = 0.f;
    for (int t = 0; t < NT; ++t) { e[t] = e[t] / sum * pmv[t]; s2 += e[t]; }
    s2 = fmaxf(s2, 1e-8f);
    float o = 0.f;
    for (int t = 0; t < NT; ++t)
        o += e[t] * VT[((size_t)t * (LDIM * LDIM) + ij) * 64 + tid];
    OUT[(size_t)ij * 64 + tid] = o / s2;
}

// ============================================================================
// Host side
// ============================================================================
struct AttnP { const float *bias_w,*g_b,*g_w,*k_w,*ln_b,*ln_g,*out_b,*out_w,*q_w,*v_w; };
struct MulP  { const float *ag_b,*ag_w,*ap_b,*ap_w,*bg_b,*bg_w,*bp_b,*bp_w,*g_b,*g_w,
                           *lnin_b,*lnin_g,*lnout_b,*lnout_g,*out_b,*out_w; };
struct TransP{ const float *l1_b,*l1_w,*l2_b,*l2_w,*ln_b,*ln_g; };
struct BlockP{ AttnP attn_e, attn_s; MulP mul_i, mul_o; TransP trans; };

extern "C" void kernel_launch(void* const* d_in, const int* in_sizes, int n_in,
                              void* d_out, int out_size, void* d_ws, size_t ws_size,
                              hipStream_t stream)
{
    (void)in_sizes; (void)n_in; (void)out_size; (void)ws_size;
    const float* m_in   = (const float*)d_in[0];   // [1,128,128,256]
    const float* z_in   = (const float*)d_in[1];   // [1,128,128,128]
    const float* angf   = (const float*)d_in[2];   // [1,4,128,51]
    const float* pairf  = (const float*)d_in[3];   // [1,4,128,128,88]
    const float* tmask  = (const float*)d_in[4];   // [1,4,128]

    // ---- param leaves (JAX tree_flatten: dict keys sorted) ----
    int pi = 5;
    auto nx = [&]() { return (const float*)d_in[pi++]; };
    const float *angle1_b = nx(), *angle1_w = nx();
    const float *angle2_b = nx(), *angle2_w = nx();
    const float *proj_b   = nx(), *proj_w   = nx();
    const float *pwa_k_w  = nx();
    const float *pwa_lnt_b = nx(), *pwa_lnt_g = nx();
    const float *pwa_lnz_b = nx(), *pwa_lnz_g = nx();
    const float *pwa_out_b = nx(), *pwa_out_w = nx();
    const float *pwa_q_w  = nx(), *pwa_v_w = nx();
    BlockP blk[2];
    for (int bi = 0; bi < 2; ++bi) {
        auto rdA = [&](AttnP& A){ A.bias_w=nx(); A.g_b=nx(); A.g_w=nx(); A.k_w=nx();
                                  A.ln_b=nx(); A.ln_g=nx(); A.out_b=nx(); A.out_w=nx();
                                  A.q_w=nx(); A.v_w=nx(); };
        auto rdM = [&](MulP& M){ M.ag_b=nx(); M.ag_w=nx(); M.ap_b=nx(); M.ap_w=nx();
                                 M.bg_b=nx(); M.bg_w=nx(); M.bp_b=nx(); M.bp_w=nx();
                                 M.g_b=nx(); M.g_w=nx(); M.lnin_b=nx(); M.lnin_g=nx();
                                 M.lnout_b=nx(); M.lnout_g=nx(); M.out_b=nx(); M.out_w=nx(); };
        rdA(blk[bi].attn_e); rdA(blk[bi].attn_s);
        rdM(blk[bi].mul_i);  rdM(blk[bi].mul_o);
        TransP& T = blk[bi].trans;
        T.l1_b=nx(); T.l1_w=nx(); T.l2_b=nx(); T.l2_w=nx(); T.ln_b=nx(); T.ln_g=nx();
    }
    const float *fln_b = nx(), *fln_g = nx();

    // ---- workspace layout ----
    const size_t RR = (size_t)NT * LDIM * LDIM;       // 65536 pair rows
    size_t off = 0;
    auto walloc = [&](size_t elems){ float* p = (float*)d_ws + off; off += elems; return p; };
    float* T_BUF = walloc(RR * CT);
    float* ZN    = walloc(RR * CT);
    float* B1    = walloc(RR * CT);   // B1+B2 contiguous: transition hidden [RR,128]
    float* B2    = walloc(RR * CT);
    float* B3    = walloc(RR * CT);
    float* B4    = walloc(RR * CT);
    float* B5    = walloc(RR * CT);
    float* TBb   = walloc(RR * NH);
    float* PMb   = walloc(RR);

    auto gemm = [&](const float* X, int ldx, const float* W, int ldw, const float* bias,
                    float* Y, int ldy, int R, int K, int N, int act,
                    const float* rowscale, const float* ewmul, int ldew,
                    int accum, int tstore) {
        int tiles = ((R + 15) / 16) * ((N + 63) / 64);
        gemm_f32_wmma_kernel<<<tiles, 128, 0, stream>>>(X, ldx, W, ldw, bias, Y, ldy,
                                                        R, K, N, act, rowscale,
                                                        ewmul, ldew, accum, tstore, LDIM);
    };
    auto ln = [&](const float* X, const float* g, const float* b, float* Y,
                  int C, int rows, int tread) {
        layernorm_kernel<<<rows, C, 0, stream>>>(X, g, b, Y, C, tread, LDIM);
    };
    auto ewmul = [&](const float* a, const float* b, const float* rs, float* o, size_t tot) {
        ewmul_kernel<<<(int)((tot + 255) / 256), 256, 0, stream>>>(a, b, rs, o, CT, tot);
    };

    // ================= m-update =================
    hipMemcpyAsync(d_out, m_in, (size_t)128 * 128 * 256 * sizeof(float),
                   hipMemcpyDeviceToDevice, stream);
    gemm(angf, 51, angle1_w, 256, angle1_b, B3, 256, 512, 51, 256, /*relu*/1,
         nullptr, nullptr, 0, 0, 0);
    float* mrows = (float*)d_out + (size_t)128 * 128 * 256;
    gemm(B3, 256, angle2_w, 256, angle2_b, mrows, 256, 512, 256, 256, 0,
         tmask /*per-row mask*/, nullptr, 0, 0, 0);

    // ================= z-update: pair stack =================
    pair_mask_kernel<<<(int)((RR + 255) / 256), 256, 0, stream>>>(tmask, PMb, (int)RR);
    gemm(pairf, 88, proj_w, CT, proj_b, T_BUF, CT, (int)RR, 88, CT, 0,
         nullptr, nullptr, 0, 0, 0);

    auto tri_attn = [&](const AttnP& A, int ending) {
        ln(T_BUF, A.ln_g, A.ln_b, ZN, CT, (int)RR, ending);
        gemm(ZN, CT, A.q_w, CT, nullptr, B1, CT, (int)RR, CT, CT, 0, nullptr, nullptr, 0, 0, 0);
        gemm(ZN, CT, A.k_w, CT, nullptr, B2, CT, (int)RR, CT, CT, 0, nullptr, nullptr, 0, 0, 0);
        gemm(ZN, CT, A.v_w, CT, nullptr, B3, CT, (int)RR, CT, CT, 0, nullptr, nullptr, 0, 0, 0);
        gemm(ZN, CT, A.bias_w, NH, nullptr, TBb, NH, (int)RR, CT, NH, 0, nullptr, nullptr, 0, 0, 0);
        gemm(ZN, CT, A.g_w, CT, A.g_b, B4, CT, (int)RR, CT, CT, /*sigmoid*/2, nullptr, nullptr, 0, 0, 0);
        tri_attn_kernel<<<NT * LDIM * NH * (LDIM / 16), 32, 0, stream>>>(B1, B2, B3, TBb, PMb, B5);
        ewmul(B4, B5, nullptr, B1, RR * CT);                      // g * o
        gemm(B1, CT, A.out_w, CT, A.out_b, T_BUF, CT, (int)RR, CT, CT, 0,
             nullptr, nullptr, 0, /*accum*/1, /*tstore*/ending);
    };
    auto tri_mul = [&](const MulP& M, int incoming) {
        ln(T_BUF, M.lnin_g, M.lnin_b, ZN, CT, (int)RR, 0);
        gemm(ZN, CT, M.ag_w, CT, M.ag_b, B1, CT, (int)RR, CT, CT, 2, nullptr, nullptr, 0, 0, 0);
        gemm(ZN, CT, M.ap_w, CT, M.ap_b, B2, CT, (int)RR, CT, CT, 0, nullptr, nullptr, 0, 0, 0);
        ewmul(B1, B2, PMb, B1, RR * CT);                          // a
        gemm(ZN, CT, M.bg_w, CT, M.bg_b, B2, CT, (int)RR, CT, CT, 2, nullptr, nullptr, 0, 0, 0);
        gemm(ZN, CT, M.bp_w, CT, M.bp_b, B3, CT, (int)RR, CT, CT, 0, nullptr, nullptr, 0, 0, 0);
        ewmul(B2, B3, PMb, B2, RR * CT);                          // b
        tri_mul_einsum_kernel<<<NT * CT * (LDIM / 16) * (LDIM / 16), 32, 0, stream>>>(B1, B2, B3, incoming);
        gemm(ZN, CT, M.g_w, CT, M.g_b, B4, CT, (int)RR, CT, CT, 2, nullptr, nullptr, 0, 0, 0);
        ln(B3, M.lnout_g, M.lnout_b, B5, CT, (int)RR, 0);
        gemm(B5, CT, M.out_w, CT, M.out_b, T_BUF, CT, (int)RR, CT, CT, 0,
             nullptr, /*ewmul=*/B4, CT, /*accum*/1, 0);
    };

    for (int bi = 0; bi < 2; ++bi) {
        tri_attn(blk[bi].attn_s, 0);
        tri_attn(blk[bi].attn_e, 1);
        tri_mul(blk[bi].mul_o, 0);
        tri_mul(blk[bi].mul_i, 1);
        const TransP& Tr = blk[bi].trans;
        ln(T_BUF, Tr.ln_g, Tr.ln_b, ZN, CT, (int)RR, 0);
        gemm(ZN, CT, Tr.l1_w, 2 * CT, Tr.l1_b, B1 /*spans B1+B2*/, 2 * CT,
             (int)RR, CT, 2 * CT, /*relu*/1, nullptr, nullptr, 0, 0, 0);
        gemm(B1, 2 * CT, Tr.l2_w, CT, Tr.l2_b, T_BUF, CT, (int)RR, 2 * CT, CT, 0,
             PMb, nullptr, 0, /*accum*/1, 0);
    }
    ln(T_BUF, fln_g, fln_b, B1, CT, (int)RR, 0);                  // final LN -> t

    // ================= pointwise attention into z =================
    ln(z_in, pwa_lnz_g, pwa_lnz_b, ZN, 128, LDIM * LDIM, 0);      // zn
    ln(B1, pwa_lnt_g, pwa_lnt_b, B2, CT, (int)RR, 0);             // tn
    gemm(ZN, 128, pwa_q_w, CT, nullptr, B3, CT, LDIM * LDIM, 128, CT, 0, nullptr, nullptr, 0, 0, 0);
    gemm(B2, CT, pwa_k_w, CT, nullptr, B4, CT, (int)RR, CT, CT, 0, nullptr, nullptr, 0, 0, 0);
    gemm(B2, CT, pwa_v_w, CT, nullptr, B5, CT, (int)RR, CT, CT, 0, nullptr, nullptr, 0, 0, 0);
    pwa_attn_kernel<<<LDIM * LDIM, 64, 0, stream>>>(B3, B4, B5, tmask, ZN);
    float* zout = (float*)d_out + (size_t)(128 + NT) * 128 * 256;
    hipMemcpyAsync(zout, z_in, (size_t)LDIM * LDIM * 128 * sizeof(float),
                   hipMemcpyDeviceToDevice, stream);
    gemm(ZN, CT, pwa_out_w, 128, pwa_out_b, zout, 128, LDIM * LDIM, CT, 128, 0,
         nullptr, nullptr, 0, /*accum: z residual*/1, 0);
}
